// TriangleMultiplicationOutgoing_41961830482535
// MI455X (gfx1250) — matrix operand order repaired and checked by reading, not compiled, not run on hardware
//
#include <hip/hip_runtime.h>

typedef __bf16 bf16;
typedef __attribute__((ext_vector_type(8)))  bf16  v8bf;
typedef __attribute__((ext_vector_type(16))) bf16  v16bf;
typedef __attribute__((ext_vector_type(8)))  float v8f;

#define NSEQ 512
#define NPOS 262144LL   // 512*512
#define CDIM 128

__device__ __forceinline__ v8f wmma_bf16(v16bf a, v16bf b, v8f c) {
  // D = A(16x32 bf16) x B(32x16 bf16) + C(16x16 f32)
  return __builtin_amdgcn_wmma_f32_16x16x32_bf16(false, a, false, b, (short)0, c, false, false);
}

// fast sigmoid: v_exp_f32 + v_rcp_f32 (avoids the IEEE-divide expansion)
__device__ __forceinline__ float sigf(float x) {
  return __builtin_amdgcn_rcpf(1.0f + __expf(-x));
}

// A fragment from a row-major bf16 tile with row stride 128 (LDS or global).
// lane l<16: row l, K = kk+{0..7} and kk+16+{0..7}; lane l>=16: +8 offset.
__device__ __forceinline__ v16bf load_a128(const bf16* x, int l, int kk) {
  const int row = l & 15;
  const int ko  = (l >> 4) << 3;
  union { v16bf v; v8bf h[2]; } u;
  u.h[0] = *(const v8bf*)(x + row * 128 + kk + ko);
  u.h[1] = *(const v8bf*)(x + row * 128 + kk + 16 + ko);
  return u.v;
}

// B fragment: lane l = column n0+(l&15) of a row-major [K][N] matrix; K = kk+{0..7} and kk+16+{0..7}; lane l>=16: +8 offset.
__device__ __forceinline__ v16bf load_b(const bf16* W, int l, int kk, int n0, int stride) {
  const int n  = n0 + (l & 15);
  const int ko = (l >> 4) << 3;
  v16bf b;
#pragma unroll
  for (int s = 0; s < 8; ++s) {
    b[s]     = W[(long long)(kk + ko + s) * stride + n];
    b[s + 8] = W[(long long)(kk + 16 + ko + s) * stride + n];
  }
  return b;
}

// ---------------- Kernel 0: convert weights f32 -> bf16 -------------------
__global__ __launch_bounds__(256)
void tri41961_wconv(const float* __restrict__ wl, const float* __restrict__ wgl,
                    const float* __restrict__ wr, const float* __restrict__ wgr,
                    const float* __restrict__ wgate, const float* __restrict__ wout,
                    bf16* __restrict__ wbf) {
  const int i = blockIdx.x * 256 + threadIdx.x;   // 0..16383
  wbf[i]             = (bf16)wl[i];
  wbf[16384 + i]     = (bf16)wgl[i];
  wbf[2 * 16384 + i] = (bf16)wr[i];
  wbf[3 * 16384 + i] = (bf16)wgr[i];
  wbf[4 * 16384 + i] = (bf16)wgate[i];
  wbf[5 * 16384 + i] = (bf16)wout[i];
}

// ------- Kernel 1: fused input LayerNorm + 5 projections (WMMA) -----------
// 16 positions per workgroup, 128 threads (4 waves).
// wave0/1: left (wl,wgl) col-halves; wave2/3: right (wr,wgr); all: 2 gate tiles.
__global__ __launch_bounds__(128)
void tri41961_proj(const float* __restrict__ act, const float* __restrict__ mask,
                   const float* __restrict__ lng, const float* __restrict__ lnb,
                   const float* __restrict__ bl, const float* __restrict__ br,
                   const float* __restrict__ bgl, const float* __restrict__ bgr,
                   const float* __restrict__ bgate,
                   const bf16* __restrict__ wbf,
                   bf16* __restrict__ Lt,     // [c][i*512+k]
                   bf16* __restrict__ RtT,    // [c][k][j]
                   bf16* __restrict__ gate) { // [p][c]
  __shared__ float xt[16 * 128];
  __shared__ bf16  xln[16 * 128];
  __shared__ float red[16 * 8 * 2];
  __shared__ float stat[16 * 2];

  const int t = threadIdx.x;
  const long long p0 = (long long)blockIdx.x * 16;

  { // load 16x128 f32 activation tile
    const float4* src = (const float4*)(act + p0 * 128);
    float4* dst = (float4*)xt;
#pragma unroll
    for (int i = 0; i < 4; ++i) dst[t + i * 128] = src[t + i * 128];
  }
  __syncthreads();

  // ---- LayerNorm over C=128 (8 threads per row) ----
  const int r = t >> 3, sg = t & 7;
  {
    float s = 0.f, s2 = 0.f;
#pragma unroll
    for (int i = 0; i < 16; ++i) {
      const float v = xt[r * 128 + sg * 16 + i];
      s += v; s2 += v * v;
    }
    red[(r * 8 + sg) * 2 + 0] = s;
    red[(r * 8 + sg) * 2 + 1] = s2;
  }
  __syncthreads();
  if (t < 16) {
    float S = 0.f, S2 = 0.f;
#pragma unroll
    for (int i = 0; i < 8; ++i) { S += red[(t * 8 + i) * 2]; S2 += red[(t * 8 + i) * 2 + 1]; }
    const float mu = S * (1.f / 128.f);
    const float var = S2 * (1.f / 128.f) - mu * mu;
    stat[t * 2 + 0] = mu;
    stat[t * 2 + 1] = rsqrtf(var + 1e-5f);
  }
  __syncthreads();
  {
    const float mu = stat[r * 2], rs = stat[r * 2 + 1];
#pragma unroll
    for (int i = 0; i < 16; ++i) {
      const int c = sg * 16 + i;
      xln[r * 128 + c] = (bf16)((xt[r * 128 + c] - mu) * rs * lng[c] + lnb[c]);
    }
  }
  __syncthreads();

  const int w = t >> 5, l = t & 31;
  const int half = w & 1, side = w >> 1;            // side 0=left, 1=right
  const bf16* Wp = wbf + side * 2 * 16384;          // wl or wr
  const bf16* Wg = Wp + 16384;                      // wgl or wgr
  const float* bp = side ? br : bl;
  const float* bg = side ? bgr : bgl;

#pragma unroll
  for (int ti = 0; ti < 4; ++ti) {
    const int n0 = (half * 4 + ti) * 16;
    v8f accP = {}; v8f accG = {};
#pragma unroll
    for (int kk = 0; kk < 128; kk += 32) {
      const v16bf a = load_a128(xln, l, kk);
      accP = wmma_bf16(a, load_b(Wp, l, kk, n0, 128), accP);
      accG = wmma_bf16(a, load_b(Wg, l, kk, n0, 128), accG);
    }
    const int col = n0 + (l & 15);
    const float bpv = bp[col], bgv = bg[col];
#pragma unroll
    for (int v = 0; v < 8; ++v) {
      const int row = v + ((l >> 4) << 3);
      const long long p = p0 + row;
      const float val = (accP[v] + bpv) * sigf(accG[v] + bgv) * mask[p];
      if (side == 0) {
        Lt[(long long)col * NPOS + p] = (bf16)val;           // [c][i][k], p=i*512+k
      } else {
        const long long j = p >> 9, kpos = p & 511;
        RtT[(long long)col * NPOS + kpos * NSEQ + j] = (bf16)val;  // [c][k][j]
      }
    }
  }

  // gate = x @ wgate + bgate (stored bf16, row-major [p][c])
  const bf16* Wgt = wbf + 4 * 16384;
#pragma unroll
  for (int ti = 0; ti < 2; ++ti) {
    const int n0 = (w * 2 + ti) * 16;
    v8f acc = {};
#pragma unroll
    for (int kk = 0; kk < 128; kk += 32)
      acc = wmma_bf16(load_a128(xln, l, kk), load_b(Wgt, l, kk, n0, 128), acc);
    const int col = n0 + (l & 15);
    const float bv = bgate[col];
#pragma unroll
    for (int v = 0; v < 8; ++v) {
      const int row = v + ((l >> 4) << 3);
      gate[(p0 + row) * 128 + col] = (bf16)(acc[v] + bv);
    }
  }
}

// ------- Kernel 2: triangle einsum, per-channel GEMM (WMMA) ---------------
// out_cm[c][i][j] = sum_k Lt[c][i][k] * RtT[c][k][j]
// grid (8,4,128): 128x64 output tile per WG; 4 waves, each a 32x64 strip
// (two A fragments share the four B fragments -> 8 WMMA per 4 b128 A loads + 64 bf16 B loads).
__global__ __launch_bounds__(128)
void tri41961_einsum(const bf16* __restrict__ Lt, const bf16* __restrict__ RtT,
                     float* __restrict__ outcm) {
  const int c  = blockIdx.z;
  const int j0 = blockIdx.x * 64;
  const int i0 = blockIdx.y * 128;
  const int t = threadIdx.x, w = t >> 5, l = t & 31;

  const bf16* Ac = Lt  + (long long)c * NPOS;  // [i][k] row-major
  const bf16* Bc = RtT + (long long)c * NPOS;  // [k][j] row-major

  const int arow0 = i0 + w * 32 + (l & 15);    // strip 0 rows
  const int arow1 = arow0 + 16;                // strip 1 rows
  const int ko    = (l >> 4) << 3;

  v8f acc[2][4] = {};
  for (int kk = 0; kk < 512; kk += 32) {
    union { v16bf v; v8bf h[2]; } a0, a1;
    a0.h[0] = *(const v8bf*)(Ac + (long long)arow0 * NSEQ + kk + ko);
    a0.h[1] = *(const v8bf*)(Ac + (long long)arow0 * NSEQ + kk + 16 + ko);
    a1.h[0] = *(const v8bf*)(Ac + (long long)arow1 * NSEQ + kk + ko);
    a1.h[1] = *(const v8bf*)(Ac + (long long)arow1 * NSEQ + kk + 16 + ko);
#pragma unroll
    for (int q = 0; q < 4; ++q) {
      const v16bf b = load_b(Bc, l, kk, j0 + 16 * q, NSEQ);
      acc[0][q] = wmma_bf16(a0.v, b, acc[0][q]);
      acc[1][q] = wmma_bf16(a1.v, b, acc[1][q]);
    }
  }

  float* oc = outcm + (long long)c * NPOS;
  const int m = (l >> 4) << 3;
  const int jj = l & 15;
#pragma unroll
  for (int s = 0; s < 2; ++s) {
#pragma unroll
    for (int q = 0; q < 4; ++q) {
#pragma unroll
      for (int v = 0; v < 8; ++v) {
        const long long i = i0 + w * 32 + s * 16 + m + v;
        oc[i * NSEQ + j0 + 16 * q + jj] = acc[s][q][v];
      }
    }
  }
}

// ------- Kernel 3: center LayerNorm + wout projection + sigmoid gate ------
__global__ __launch_bounds__(128)
void tri41961_final(const float* __restrict__ outcm, const bf16* __restrict__ gate,
                    const bf16* __restrict__ wbf,
                    const float* __restrict__ lncg, const float* __restrict__ lncb,
                    const float* __restrict__ bout, float* __restrict__ out) {
  __shared__ float xt[16 * 128];
  __shared__ bf16  xln[16 * 128];
  __shared__ float red[16 * 8 * 2];
  __shared__ float stat[16 * 2];

  const int t = threadIdx.x;
  const long long p0 = (long long)blockIdx.x * 16;

  { // gather 16 positions x 128 channels from channel-major layout
    const float* src = outcm + (long long)t * NPOS + p0;
#pragma unroll
    for (int q = 0; q < 4; ++q) {
      const float4 v = *(const float4*)(src + q * 4);
      xt[(q * 4 + 0) * 128 + t] = v.x;
      xt[(q * 4 + 1) * 128 + t] = v.y;
      xt[(q * 4 + 2) * 128 + t] = v.z;
      xt[(q * 4 + 3) * 128 + t] = v.w;
    }
  }
  __syncthreads();

  const int r = t >> 3, sg = t & 7;
  {
    float s = 0.f, s2 = 0.f;
#pragma unroll
    for (int i = 0; i < 16; ++i) {
      const float v = xt[r * 128 + sg * 16 + i];
      s += v; s2 += v * v;
    }
    red[(r * 8 + sg) * 2 + 0] = s;
    red[(r * 8 + sg) * 2 + 1] = s2;
  }
  __syncthreads();
  if (t < 16) {
    float S = 0.f, S2 = 0.f;
#pragma unroll
    for (int i = 0; i < 8; ++i) { S += red[(t * 8 + i) * 2]; S2 += red[(t * 8 + i) * 2 + 1]; }
    const float mu = S * (1.f / 128.f);
    const float var = S2 * (1.f / 128.f) - mu * mu;
    stat[t * 2 + 0] = mu;
    stat[t * 2 + 1] = rsqrtf(var + 1e-5f);
  }
  __syncthreads();
  {
    const float mu = stat[r * 2], rs = stat[r * 2 + 1];
#pragma unroll
    for (int i = 0; i < 16; ++i) {
      const int c = sg * 16 + i;
      xln[r * 128 + c] = (bf16)((xt[r * 128 + c] - mu) * rs * lncg[c] + lncb[c]);
    }
  }
  __syncthreads();

  const int w = t >> 5, l = t & 31;
  const bf16* Wout = wbf + 5 * 16384;
#pragma unroll
  for (int ti = 0; ti < 2; ++ti) {
    const int n0 = (w * 2 + ti) * 16;
    v8f acc = {};
#pragma unroll
    for (int kk = 0; kk < 128; kk += 32)
      acc = wmma_bf16(load_a128(xln, l, kk), load_b(Wout, l, kk, n0, 128), acc);
    const int col = n0 + (l & 15);
    const float bv = bout[col];
#pragma unroll
    for (int v = 0; v < 8; ++v) {
      const int row = v + ((l >> 4) << 3);
      const long long p = p0 + row;
      const float g = (float)gate[p * 128 + col];
      out[p * 128 + col] = (acc[v] + bv) * sigf(g);
    }
  }
}

extern "C" void kernel_launch(void* const* d_in, const int* in_sizes, int n_in,
                              void* d_out, int out_size, void* d_ws, size_t ws_size,
                              hipStream_t stream) {
  const float* act   = (const float*)d_in[0];
  const float* mask  = (const float*)d_in[1];
  const float* lng   = (const float*)d_in[2];
  const float* lnb   = (const float*)d_in[3];
  const float* wl    = (const float*)d_in[4];
  const float* bl    = (const float*)d_in[5];
  const float* wr    = (const float*)d_in[6];
  const float* br    = (const float*)d_in[7];
  const float* wgl   = (const float*)d_in[8];
  const float* bgl   = (const float*)d_in[9];
  const float* wgr   = (const float*)d_in[10];
  const float* bgr   = (const float*)d_in[11];
  const float* wgate = (const float*)d_in[12];
  const float* bgate = (const float*)d_in[13];
  const float* lncg  = (const float*)d_in[14];
  const float* lncb  = (const float*)d_in[15];
  const float* wout  = (const float*)d_in[16];
  const float* bout  = (const float*)d_in[17];
  float* out = (float*)d_out;

  char* ws = (char*)d_ws;
  bf16*  Lt    = (bf16*)(ws + 0);            //  64 MB  bf16 [c][i][k]
  bf16*  RtT   = (bf16*)(ws + 67108864);     //  64 MB  bf16 [c][k][j]
  bf16*  gate  = (bf16*)(ws + 134217728);    //  64 MB  bf16 [p][c]
  float* outcm = (float*)(ws + 201326592);   // 128 MB  f32  [c][p]
  bf16*  wbf   = (bf16*)(ws + 335544320);    // 192 KB  bf16 weights x6

  tri41961_wconv<<<64, 256, 0, stream>>>(wl, wgl, wr, wgr, wgate, wout, wbf);
  tri41961_proj<<<16384, 128, 0, stream>>>(act, mask, lng, lnb, bl, br, bgl, bgr,
                                           bgate, wbf, Lt, RtT, gate);
  tri41961_einsum<<<dim3(8, 4, 128), 128, 0, stream>>>(Lt, RtT, outcm);
  tri41961_final<<<16384, 128, 0, stream>>>(outcm, gate, wbf, lncg, lncb, bout, out);
}
